// DynamicLoRALinear_38852274159666
// MI455X (gfx1250) — compile-verified, matrix operand order checked by reading, not run on hardware
//
#include <hip/hip_runtime.h>

#define IN_F   4096
#define OUT_F  4096
#define RANK   16
#define MTOT   8192           // B*T = 8*1024
#define SCALING 2.0f          // ALPHA/R = 32/16
#define LDK    40             // LDS row stride in halves (80B: 16B-aligned, conflict-padded)

typedef __attribute__((ext_vector_type(16))) __bf16          v16bf_t;
typedef __attribute__((ext_vector_type(8)))  float           v8f_t;
typedef __attribute__((ext_vector_type(4)))  unsigned short  u16x4;
typedef __attribute__((ext_vector_type(8)))  unsigned short  u16x8;

union FragU { u16x8 h[2]; v16bf_t v; };

__device__ __forceinline__ v8f_t zero8() {
    v8f_t z = {0.f, 0.f, 0.f, 0.f, 0.f, 0.f, 0.f, 0.f};
    return z;
}

// round-to-nearest-even f32 -> bf16 bits
__device__ __forceinline__ unsigned short bf16_rne(float x) {
    unsigned int u = __float_as_uint(x);
    unsigned int r = ((u >> 16) & 1u) + 0x7fffu;
    return (unsigned short)((u + r) >> 16);
}

// split f32 into truncated-hi bf16 + RNE residual-lo bf16 (bf16x3 scheme)
__device__ __forceinline__ void bf16_split(float x, unsigned short& hi, unsigned short& lo) {
    unsigned int u  = __float_as_uint(x);
    unsigned int hb = u & 0xffff0000u;
    hi = (unsigned short)(hb >> 16);
    lo = bf16_rne(x - __uint_as_float(hb));
}

__device__ __forceinline__ void split4(const float4 v, u16x4& hv, u16x4& lv) {
    unsigned short h0, l0, h1, l1, h2, l2, h3, l3;
    bf16_split(v.x, h0, l0);
    bf16_split(v.y, h1, l1);
    bf16_split(v.z, h2, l2);
    bf16_split(v.w, h3, l3);
    hv[0] = h0; hv[1] = h1; hv[2] = h2; hv[3] = h3;
    lv[0] = l0; lv[1] = l1; lv[2] = l2; lv[3] = l3;
}

// Build a 16x32 / 32x16 bf16 WMMA fragment from a K-contiguous LDS tile.
// ISA layout: lanes<16 hold K{khalf..+7} in VGPR0-3, K{khalf+16..+23} in VGPR4-7
// (khalf=0); lanes>=16 use khalf=8.
__device__ __forceinline__ v16bf_t ld_frag(const unsigned short* p, int rowHalves, int khalf) {
    FragU f;
    f.h[0] = *(const u16x8*)(p + rowHalves + khalf);
    f.h[1] = *(const u16x8*)(p + rowHalves + khalf + 16);
    return f.v;
}

#define WMMA_BF16(A, B, C) \
    __builtin_amdgcn_wmma_f32_16x16x32_bf16(false, (A), false, (B), (short)0, (C), false, false)

// ---------------------------------------------------------------------------
// Pack pass: fp32 -> (hi bf16, lo bf16) arrays, one-time, memory-bound.
// Removes all conversion VALU from the GEMM K-loop.
// ---------------------------------------------------------------------------
__global__ __launch_bounds__(256) void pack_bf16_kernel(
    const float* __restrict__ src,
    unsigned short* __restrict__ hi, unsigned short* __restrict__ lo,
    long n4)
{
    long stride = (long)gridDim.x * blockDim.x;
    for (long i = (long)blockIdx.x * blockDim.x + threadIdx.x; i < n4; i += stride) {
        const float4 v = *(const float4*)(src + i * 4);
        u16x4 hv, lv;
        split4(v, hv, lv);
        *(u16x4*)(hi + i * 4) = hv;
        *(u16x4*)(lo + i * 4) = lv;
    }
}

// ---------------------------------------------------------------------------
// Kernel 1: h[m, r] = sum_k x[m,k] * A_s[k,r]  (rank-16, bf16x3 WMMA)
// ---------------------------------------------------------------------------
__global__ __launch_bounds__(256) void lora_h_kernel(
    const float* __restrict__ x, const int* __restrict__ amap,
    const float* __restrict__ As, float* __restrict__ h)
{
    __shared__ __align__(16) unsigned short Ahi[128 * LDK], Alo[128 * LDK];
    __shared__ __align__(16) unsigned short Bhi[RANK * LDK], Blo[RANK * LDK];

    const int t     = threadIdx.x;
    const int lane  = t & 31;
    const int wid   = t >> 5;
    const int laneM = lane & 15;
    const int khalf = (lane & 16) ? 8 : 0;
    const int m0    = blockIdx.x * 128;          // 1024 % 128 == 0 -> uniform batch
    const int s     = amap[m0 >> 10];
    const float* Abase = As + (size_t)s * IN_F * RANK;

    v8f_t acc = zero8();

    for (int k0 = 0; k0 < IN_F; k0 += 32) {
        #pragma unroll
        for (int j = 0; j < 4; ++j) {
            int idx = j * 256 + t;
            int row = idx >> 3, c4 = idx & 7;
            const float4 v = *(const float4*)(x + (size_t)(m0 + row) * IN_F + k0 + c4 * 4);
            u16x4 hv, lv;
            split4(v, hv, lv);
            *(u16x4*)&Ahi[row * LDK + c4 * 4] = hv;
            *(u16x4*)&Alo[row * LDK + c4 * 4] = lv;
        }
        #pragma unroll
        for (int j = 0; j < 2; ++j) {
            int idx = j * 256 + t;
            int k = idx >> 4, r = idx & 15;
            unsigned short hv, lv;
            bf16_split(Abase[(size_t)(k0 + k) * RANK + r], hv, lv);
            Bhi[r * LDK + k] = hv;
            Blo[r * LDK + k] = lv;
        }
        __syncthreads();

        v16bf_t ah = ld_frag(Ahi, (wid * 16 + laneM) * LDK, khalf);
        v16bf_t al = ld_frag(Alo, (wid * 16 + laneM) * LDK, khalf);
        v16bf_t bh = ld_frag(Bhi, laneM * LDK, khalf);
        v16bf_t bl = ld_frag(Blo, laneM * LDK, khalf);
        acc = WMMA_BF16(ah, bh, acc);
        acc = WMMA_BF16(al, bh, acc);
        acc = WMMA_BF16(ah, bl, acc);
        __syncthreads();
    }

    const int rbase = (lane & 16) ? 8 : 0;
    #pragma unroll
    for (int v = 0; v < 8; ++v)
        h[(size_t)(m0 + wid * 16 + rbase + v) * RANK + laneM] = acc[v];
}

// ---------------------------------------------------------------------------
// Kernel 2 (fast path): GEMM on pre-packed hi/lo bf16. WG tile 128x128,
// 8 waves 4(M)x2(N), wave tile 32x64 -> 24 WMMAs/wave/K-step, staging is
// pure b128 copies (zero conversion VALU in the loop).
// ---------------------------------------------------------------------------
__global__ __launch_bounds__(256) void gemm_packed_kernel(
    const unsigned short* __restrict__ xhi, const unsigned short* __restrict__ xlo,
    const unsigned short* __restrict__ whi, const unsigned short* __restrict__ wlo,
    const int* __restrict__ amap, const float* __restrict__ bias,
    const float* __restrict__ Bs, const float* __restrict__ h,
    float* __restrict__ out)
{
    __shared__ __align__(16) unsigned short Ahi[128 * LDK], Alo[128 * LDK];
    __shared__ __align__(16) unsigned short Bhi[128 * LDK], Blo[128 * LDK];

    const int t     = threadIdx.x;
    const int lane  = t & 31;
    const int wid   = t >> 5;
    const int laneM = lane & 15;
    const int khalf = (lane & 16) ? 8 : 0;
    const int m0    = blockIdx.y * 128;
    const int n0    = blockIdx.x * 128;
    const int m_off = (wid & 3) * 32;
    const int n_off = (wid >> 2) * 64;

    v8f_t acc[2][4];
    #pragma unroll
    for (int mi = 0; mi < 2; ++mi)
        #pragma unroll
        for (int ni = 0; ni < 4; ++ni)
            acc[mi][ni] = zero8();

    for (int k0 = 0; k0 < IN_F; k0 += 32) {
        // stage 128x32 x-tile and 128x32 W-tile (hi+lo): pure b128 copies
        #pragma unroll
        for (int j = 0; j < 2; ++j) {
            int c = j * 256 + t;            // 0..511 chunks of 8 halves
            int row = c >> 2, c8 = (c & 3) * 8;
            size_t gx = (size_t)(m0 + row) * IN_F + k0 + c8;
            size_t gw = (size_t)(n0 + row) * IN_F + k0 + c8;
            if (k0 + 32 < IN_F) {
                __builtin_prefetch(xhi + gx + 32, 0, 1);   // global_prefetch_b8
                __builtin_prefetch(whi + gw + 32, 0, 1);
            }
            *(u16x8*)&Ahi[row * LDK + c8] = *(const u16x8*)(xhi + gx);
            *(u16x8*)&Alo[row * LDK + c8] = *(const u16x8*)(xlo + gx);
            *(u16x8*)&Bhi[row * LDK + c8] = *(const u16x8*)(whi + gw);
            *(u16x8*)&Blo[row * LDK + c8] = *(const u16x8*)(wlo + gw);
        }
        __syncthreads();

        v16bf_t ah[2], al[2], bh[4], bl[4];
        #pragma unroll
        for (int mi = 0; mi < 2; ++mi) {
            ah[mi] = ld_frag(Ahi, (m_off + mi * 16 + laneM) * LDK, khalf);
            al[mi] = ld_frag(Alo, (m_off + mi * 16 + laneM) * LDK, khalf);
        }
        #pragma unroll
        for (int ni = 0; ni < 4; ++ni) {
            bh[ni] = ld_frag(Bhi, (n_off + ni * 16 + laneM) * LDK, khalf);
            bl[ni] = ld_frag(Blo, (n_off + ni * 16 + laneM) * LDK, khalf);
        }
        #pragma unroll
        for (int mi = 0; mi < 2; ++mi)
            #pragma unroll
            for (int ni = 0; ni < 4; ++ni) {
                acc[mi][ni] = WMMA_BF16(ah[mi], bh[ni], acc[mi][ni]);
                acc[mi][ni] = WMMA_BF16(al[mi], bh[ni], acc[mi][ni]);
                acc[mi][ni] = WMMA_BF16(ah[mi], bl[ni], acc[mi][ni]);
            }
        __syncthreads();
    }

    // ---- fused LoRA epilogue: acc += (scaling*h) @ B_s, K=16 zero-padded to 32
    {
        const int s = amap[m0 >> 10];
        {   // scaled h tile 128x16 into Ahi (K 16..31 zeroed)
            int row = t >> 1, cb = (t & 1) * 8;
            const float* hp = h + (size_t)(m0 + row) * RANK + cb;
            u16x8 hv, hz = {0, 0, 0, 0, 0, 0, 0, 0};
            #pragma unroll
            for (int e = 0; e < 8; ++e) hv[e] = bf16_rne(hp[e] * SCALING);
            *(u16x8*)&Ahi[row * LDK + cb]      = hv;
            *(u16x8*)&Ahi[row * LDK + 16 + cb] = hz;
        }
        {   // Bm^T tile into Bhi: Bhi[n][k] = B_s[s][k][n0+n] (K 16..31 zeroed)
            int n = t & 127, kb = t >> 7;   // kb in 0..1
            #pragma unroll
            for (int j = 0; j < 8; ++j) {
                int k = kb * 8 + j;
                Bhi[n * LDK + k]      = bf16_rne(Bs[((size_t)s * RANK + k) * OUT_F + n0 + n]);
                Bhi[n * LDK + 16 + k] = 0;
            }
        }
        __syncthreads();
        #pragma unroll
        for (int mi = 0; mi < 2; ++mi) {
            v16bf_t a = ld_frag(Ahi, (m_off + mi * 16 + laneM) * LDK, khalf);
            #pragma unroll
            for (int ni = 0; ni < 4; ++ni) {
                v16bf_t b = ld_frag(Bhi, (n_off + ni * 16 + laneM) * LDK, khalf);
                acc[mi][ni] = WMMA_BF16(a, b, acc[mi][ni]);
            }
        }
    }

    // ---- bias add + store (C layout: row = v + 8*(lane>=16), col = lane&15)
    const int rbase = (lane & 16) ? 8 : 0;
    #pragma unroll
    for (int mi = 0; mi < 2; ++mi) {
        int rg0 = m0 + m_off + mi * 16 + rbase;
        #pragma unroll
        for (int ni = 0; ni < 4; ++ni) {
            int cg = n0 + n_off + ni * 16 + laneM;
            float bv = bias[cg];
            #pragma unroll
            for (int v = 0; v < 8; ++v)
                out[(size_t)(rg0 + v) * OUT_F + cg] = acc[mi][ni][v] + bv;
        }
    }
}

// ---------------------------------------------------------------------------
// Kernel 2 (fallback, small ws): convert-in-loop bf16x3 GEMM, WG tile 128x64.
// ---------------------------------------------------------------------------
__global__ __launch_bounds__(256) void lora_linear_kernel(
    const float* __restrict__ x, const int* __restrict__ amap,
    const float* __restrict__ W, const float* __restrict__ bias,
    const float* __restrict__ Bs, const float* __restrict__ h,
    float* __restrict__ out)
{
    __shared__ __align__(16) unsigned short Ahi[128 * LDK], Alo[128 * LDK];
    __shared__ __align__(16) unsigned short Bhi[64 * LDK],  Blo[64 * LDK];

    const int t     = threadIdx.x;
    const int lane  = t & 31;
    const int wid   = t >> 5;
    const int laneM = lane & 15;
    const int khalf = (lane & 16) ? 8 : 0;
    const int m0    = blockIdx.y * 128;
    const int n0    = blockIdx.x * 64;
    const int m_off = (wid & 3) * 32;
    const int n_off = (wid >> 2) * 32;

    v8f_t acc[2][2];
    #pragma unroll
    for (int mi = 0; mi < 2; ++mi)
        #pragma unroll
        for (int ni = 0; ni < 2; ++ni)
            acc[mi][ni] = zero8();

    for (int k0 = 0; k0 < IN_F; k0 += 32) {
        #pragma unroll
        for (int j = 0; j < 4; ++j) {
            int idx = j * 256 + t;
            int row = idx >> 3, c4 = idx & 7;
            const float4 v = *(const float4*)(x + (size_t)(m0 + row) * IN_F + k0 + c4 * 4);
            u16x4 hv, lv;
            split4(v, hv, lv);
            *(u16x4*)&Ahi[row * LDK + c4 * 4] = hv;
            *(u16x4*)&Alo[row * LDK + c4 * 4] = lv;
        }
        #pragma unroll
        for (int j = 0; j < 2; ++j) {
            int idx = j * 256 + t;
            int n = idx >> 3, c4 = idx & 7;
            const float4 v = *(const float4*)(W + (size_t)(n0 + n) * IN_F + k0 + c4 * 4);
            u16x4 hv, lv;
            split4(v, hv, lv);
            *(u16x4*)&Bhi[n * LDK + c4 * 4] = hv;
            *(u16x4*)&Blo[n * LDK + c4 * 4] = lv;
        }
        __syncthreads();

        v16bf_t ah[2], al[2], bh[2], bl[2];
        #pragma unroll
        for (int mi = 0; mi < 2; ++mi) {
            ah[mi] = ld_frag(Ahi, (m_off + mi * 16 + laneM) * LDK, khalf);
            al[mi] = ld_frag(Alo, (m_off + mi * 16 + laneM) * LDK, khalf);
        }
        #pragma unroll
        for (int ni = 0; ni < 2; ++ni) {
            bh[ni] = ld_frag(Bhi, (n_off + ni * 16 + laneM) * LDK, khalf);
            bl[ni] = ld_frag(Blo, (n_off + ni * 16 + laneM) * LDK, khalf);
        }
        #pragma unroll
        for (int mi = 0; mi < 2; ++mi)
            #pragma unroll
            for (int ni = 0; ni < 2; ++ni) {
                acc[mi][ni] = WMMA_BF16(ah[mi], bh[ni], acc[mi][ni]);
                acc[mi][ni] = WMMA_BF16(al[mi], bh[ni], acc[mi][ni]);
                acc[mi][ni] = WMMA_BF16(ah[mi], bl[ni], acc[mi][ni]);
            }
        __syncthreads();
    }

    {
        const int s = amap[m0 >> 10];
        {
            int row = t >> 1, cb = (t & 1) * 8;
            const float* hp = h + (size_t)(m0 + row) * RANK + cb;
            u16x8 hv, hz = {0, 0, 0, 0, 0, 0, 0, 0};
            #pragma unroll
            for (int e = 0; e < 8; ++e) hv[e] = bf16_rne(hp[e] * SCALING);
            *(u16x8*)&Ahi[row * LDK + cb]      = hv;
            *(u16x8*)&Ahi[row * LDK + 16 + cb] = hz;
        }
        {
            int n = t & 63, kb = t >> 6;
            #pragma unroll
            for (int j = 0; j < 4; ++j) {
                int k = kb + j * 4;
                Bhi[n * LDK + k]      = bf16_rne(Bs[((size_t)s * RANK + k) * OUT_F + n0 + n]);
                Bhi[n * LDK + 16 + k] = 0;
            }
        }
        __syncthreads();
        #pragma unroll
        for (int mi = 0; mi < 2; ++mi) {
            v16bf_t a = ld_frag(Ahi, (m_off + mi * 16 + laneM) * LDK, khalf);
            #pragma unroll
            for (int ni = 0; ni < 2; ++ni) {
                v16bf_t b = ld_frag(Bhi, (n_off + ni * 16 + laneM) * LDK, khalf);
                acc[mi][ni] = WMMA_BF16(a, b, acc[mi][ni]);
            }
        }
    }

    const int rbase = (lane & 16) ? 8 : 0;
    #pragma unroll
    for (int mi = 0; mi < 2; ++mi) {
        int rg0 = m0 + m_off + mi * 16 + rbase;
        #pragma unroll
        for (int ni = 0; ni < 2; ++ni) {
            int cg = n0 + n_off + ni * 16 + laneM;
            float bv = bias[cg];
            #pragma unroll
            for (int v = 0; v < 8; ++v)
                out[(size_t)(rg0 + v) * OUT_F + cg] = acc[mi][ni][v] + bv;
        }
    }
}

extern "C" void kernel_launch(void* const* d_in, const int* in_sizes, int n_in,
                              void* d_out, int out_size, void* d_ws, size_t ws_size,
                              hipStream_t stream) {
    const float* x    = (const float*)d_in[0];
    const int*   amap = (const int*)d_in[1];
    const float* W    = (const float*)d_in[2];
    const float* b    = (const float*)d_in[3];
    const float* As   = (const float*)d_in[4];
    const float* Bs   = (const float*)d_in[5];
    float* out = (float*)d_out;

    const size_t hBytes = (size_t)MTOT * RANK * sizeof(float);      // 512 KB
    const size_t xElems = (size_t)MTOT * IN_F;
    const size_t wElems = (size_t)OUT_F * IN_F;
    const size_t need   = hBytes + 2 * xElems * 2 + 2 * wElems * 2; // ~192.5 MB

    float* h = (float*)d_ws;
    lora_h_kernel<<<dim3(MTOT / 128), 256, 0, stream>>>(x, amap, As, h);

    if (ws_size >= need) {
        unsigned short* xhi = (unsigned short*)((char*)d_ws + hBytes);
        unsigned short* xlo = xhi + xElems;
        unsigned short* whi = xlo + xElems;
        unsigned short* wlo = whi + wElems;
        pack_bf16_kernel<<<dim3(2048), 256, 0, stream>>>(x, xhi, xlo, (long)(xElems / 4));
        pack_bf16_kernel<<<dim3(2048), 256, 0, stream>>>(W, whi, wlo, (long)(wElems / 4));
        gemm_packed_kernel<<<dim3(OUT_F / 128, MTOT / 128), 256, 0, stream>>>(
            xhi, xlo, whi, wlo, amap, b, Bs, h, out);
    } else {
        lora_linear_kernel<<<dim3(OUT_F / 64, MTOT / 128), 256, 0, stream>>>(
            x, amap, W, b, Bs, h, out);
    }
}
